// SequenceToPairBlock_214748365011
// MI455X (gfx1250) — matrix operand order repaired and checked by reading, not compiled, not run on hardware
//
#include <hip/hip_runtime.h>
#include <hip/hip_bf16.h>
#include <math.h>

// ---- problem constants (reference: B=2,S=8192,C=1536,W=16,H=32,F=128,P=512) ----
#define BQ 2
#define SEQ 8192
#define CH 1536
#define WID 16
#define HH 32
#define FF 128
#define PP 512
#define P2 1024
#define PE_ROWS 1040      // 2P padded by 16 rows so 16-row B tiles never run OOB
#define HF 4096           // H*F

typedef __bf16 bf16;
typedef __attribute__((ext_vector_type(16))) __bf16 v16bf;
typedef __attribute__((ext_vector_type(8)))  __bf16 v8bf;
typedef __attribute__((ext_vector_type(8)))  float  v8f;

__device__ __forceinline__ bf16 to_bf16(float f) {
  unsigned u = __builtin_bit_cast(unsigned, f);
  u += 0x7fffu + ((u >> 16) & 1u);                 // round-to-nearest-even
  unsigned short h = (unsigned short)(u >> 16);
  return __builtin_bit_cast(bf16, h);
}

// D = A*B + C, bf16 inputs, f32 accum (codegen-confirmed CDNA5 builtin)
__device__ __forceinline__ v8f wmma_bf16(v16bf a, v16bf b, v8f c) {
  return __builtin_amdgcn_wmma_f32_16x16x32_bf16(false, a, false, b, (short)0, c,
                                                 false, false);
}

// A fragment: row-major [rows][ld], 16 rows x 32 K. ISA 16-bit A layout:
// lane L (r=L&15, hi=L>>4): elems0..7 = K hi*8.., elems8..15 = K 16+hi*8..
__device__ __forceinline__ v16bf load_a_frag(const bf16* base, int ld) {
  int lane = threadIdx.x & 31;
  int r = lane & 15, hi = lane >> 4;
  const bf16* p = base + (size_t)r * ld + hi * 8;
  v8bf lo = *(const v8bf*)p;
  v8bf h8 = *(const v8bf*)(p + 16);
  v16bf out;
#pragma unroll
  for (int i = 0; i < 8; ++i) { out[i] = lo[i]; out[i + 8] = h8[i]; }
  return out;
}

// B fragment from B^T storage [N][K]. ISA 16-bit B layout: lane L (n=L&15,
// hi=L>>4): elems0..7 = K hi*8.., elems8..15 = K 16+hi*8.. of column n  (== B^T row n).
__device__ __forceinline__ v16bf load_b_frag(const bf16* base, int ld) {
  int lane = threadIdx.x & 31;
  int n = lane & 15, hi = lane >> 4;
  const bf16* p = base + (size_t)n * ld + hi * 8;
  v8bf lo = *(const v8bf*)p;
  v8bf h8 = *(const v8bf*)(p + 16);
  v16bf out;
#pragma unroll
  for (int i = 0; i < 8; ++i) { out[i] = lo[i]; out[i + 8] = h8[i]; }
  return out;
}

// ---------------- 1) pool + RMSnorm + gelu ----------------
__global__ void prep_kernel(const float* __restrict__ x, const float* __restrict__ norm_w,
                            bf16* __restrict__ xn, bf16* __restrict__ xg) {
  int row = blockIdx.x;                 // b*P + p
  int b = row >> 9, p = row & 511;
  int tid = threadIdx.x;
  __shared__ float red[256];
  float vals[6];
  float ss = 0.f;
#pragma unroll
  for (int j = 0; j < 6; ++j) {
    int c = j * 256 + tid;
    const float* xp = x + ((size_t)(b * SEQ + p * WID)) * CH + c;
    float s = 0.f;
#pragma unroll
    for (int w = 0; w < WID; ++w) s += xp[(size_t)w * CH];
    s *= (1.0f / WID);
    vals[j] = s;
    ss += s * s;
  }
  red[tid] = ss;
  __syncthreads();
  for (int off = 128; off > 0; off >>= 1) {
    if (tid < off) red[tid] += red[tid + off];
    __syncthreads();
  }
  float rs = rsqrtf(red[0] * (1.0f / CH) + 1e-8f);
#pragma unroll
  for (int j = 0; j < 6; ++j) {
    int c = j * 256 + tid;
    float v = vals[j] * rs * norm_w[c];
    xn[(size_t)row * CH + c] = to_bf16(v);
    float g = 0.5f * v * (1.0f + erff(v * 0.70710678118654752f));
    xg[(size_t)row * CH + c] = to_bf16(g);
  }
}

// ---------------- 2) transpose + f32->bf16 : W[K][N] -> out[N][K] ----------------
__global__ void tconv_kernel(const float* __restrict__ W, bf16* __restrict__ out,
                             int K, int N) {
  size_t idx = (size_t)blockIdx.x * blockDim.x + threadIdx.x;
  size_t tot = (size_t)K * N;
  if (idx >= tot) return;
  int k = (int)(idx / N), n = (int)(idx % N);
  out[(size_t)n * K + k] = to_bf16(W[idx]);
}

// ---------------- 3) positional encoding [H][PE_ROWS][F] bf16 ----------------
__global__ void posenc_kernel(const float* __restrict__ Wpos, const float* __restrict__ bpos,
                              bf16* __restrict__ penc) {
  int p2 = blockIdx.x;
  int tid = threadIdx.x;
  if (p2 >= P2) {                 // zero the 16 pad rows (never read, keep clean)
#pragma unroll
    for (int cj = 0; cj < 16; ++cj) {
      int col = cj * 256 + tid;
      int h = col >> 7, f = col & 127;
      penc[((size_t)h * PE_ROWS + p2) * FF + f] = to_bf16(0.f);
    }
    return;
  }
  float relv = (float)(p2 - PP);
  float sgn = (relv > 0.f) ? 1.f : (relv < 0.f ? -1.f : 0.f);
  float ar = fabsf(relv);
  float oh[16];
  const float l2 = log2f(497.0f);   // geomspace(1, P-half+1, 16, endpoint=False)
#pragma unroll
  for (int i = 0; i < 16; ++i) {
    float cw = (float)i + exp2f(((float)i * (1.0f / 16.0f)) * l2);
    oh[i] = (cw > ar) ? 1.f : 0.f;
  }
#pragma unroll
  for (int cj = 0; cj < 16; ++cj) {
    int col = cj * 256 + tid;
    float acc = bpos[col];
#pragma unroll
    for (int i = 0; i < 16; ++i)
      acc += oh[i] * (Wpos[i * HF + col] + sgn * Wpos[(i + 16) * HF + col]);
    int h = col >> 7, f = col & 127;
    penc[((size_t)h * PE_ROWS + p2) * FF + f] = to_bf16(acc);
  }
}

// ---------------- 4) q/k projection: [1024x1536]@[1536x4096] via WMMA ----------------
// outputs: plain and +rotary-bias copies, layout [b][h][p][f] bf16
__global__ void proj_kernel(const bf16* __restrict__ A, const bf16* __restrict__ BT,
                            const float* __restrict__ rbias,
                            bf16* __restrict__ outP, bf16* __restrict__ outB) {
  int wave = threadIdx.x >> 5;
  int id = blockIdx.x * 8 + wave;       // 16384 wave-tiles (64 M x 256 N)
  int mt = id >> 8, nt = id & 255;
  v8f acc = {};
  const bf16* abase = A + (size_t)(mt * 16) * CH;
  const bf16* bbase = BT + (size_t)(nt * 16) * CH;
  for (int kk = 0; kk < CH; kk += 32)
    acc = wmma_bf16(load_a_frag(abase + kk, CH), load_b_frag(bbase + kk, CH), acc);
  int lane = threadIdx.x & 31;
  int n = lane & 15, mb = (lane >> 4) * 8;
  int col = nt * 16 + n;
  int h = col >> 7, f = col & 127;
  float rb = rbias[col];
#pragma unroll
  for (int j = 0; j < 8; ++j) {
    int row = mt * 16 + mb + j;
    int b = row >> 9, p = row & 511;
    size_t o = (((size_t)(b * HH + h) * PP) + p) * FF + f;
    outP[o] = to_bf16(acc[j]);
    outB[o] = to_bf16(acc[j] + rb);
  }
}

// ---------------- 5) y projections: [1024x1536]@[1536x128], f32 out ----------------
__global__ void ygemm_kernel(const bf16* __restrict__ A, const bf16* __restrict__ BT,
                             float* __restrict__ out) {
  int wave = threadIdx.x >> 5;
  int id = blockIdx.x * 8 + wave;       // 512 wave-tiles (64 M x 8 N)
  int mt = id >> 3, nt = id & 7;
  v8f acc = {};
  const bf16* abase = A + (size_t)(mt * 16) * CH;
  const bf16* bbase = BT + (size_t)(nt * 16) * CH;
  for (int kk = 0; kk < CH; kk += 32)
    acc = wmma_bf16(load_a_frag(abase + kk, CH), load_b_frag(bbase + kk, CH), acc);
  int lane = threadIdx.x & 31;
  int n = lane & 15, mb = (lane >> 4) * 8;
#pragma unroll
  for (int j = 0; j < 8; ++j)
    out[(size_t)(mt * 16 + mb + j) * FF + nt * 16 + n] = acc[j];
}

// ---------------- 6) fused attention + rel-shift + a@Wout + epilogue ----------------
// block = 256 thr (8 waves), one 16x16 (q,k) tile of one batch. Per wave: 4 heads.
__global__ void __launch_bounds__(256) pair_kernel(
    const bf16* __restrict__ q, const bf16* __restrict__ qb,
    const bf16* __restrict__ k, const bf16* __restrict__ kb,
    const bf16* __restrict__ penc, const bf16* __restrict__ woutT,
    const float* __restrict__ bout, const float* __restrict__ yq,
    const float* __restrict__ yk, float* __restrict__ out) {
  __shared__ bf16  aT[256 * 32];       // 16KB: a_flat [qk=256][h=32] bf16
  __shared__ float gsc[8][16 * 64];    // 32KB: per-wave [row][Gq j0..31 | Gk j0..31]
  __shared__ float qksc[8][16 * 16];   // 8KB : per-wave qk tile

  int bid = blockIdx.x;
  int b = bid >> 10;
  int t = bid & 1023;
  int qt = t >> 5, kt = t & 31;
  int q0 = qt * 16, k0 = kt * 16;
  int wave = threadIdx.x >> 5;
  int lane = threadIdx.x & 31;
  int n = lane & 15, mb = (lane >> 4) * 8;

  // rel-shift: E_q needed at p = P + (k-q); within tile p = pq0 + (kk-qq+15)
  int pq0 = PP + k0 - q0 - 15;         // in [1, 993] -> rows pq0..pq0+31 valid
  int pk0 = PP + q0 - k0 - 15;

  for (int hi4 = 0; hi4 < 4; ++hi4) {
    int h = wave * 4 + hi4;
    const bf16* qbase  = q  + (((size_t)(b * HH + h) * PP) + q0) * FF;
    const bf16* kbase  = k  + (((size_t)(b * HH + h) * PP) + k0) * FF;
    const bf16* qbbase = qb + (((size_t)(b * HH + h) * PP) + q0) * FF;
    const bf16* kbbase = kb + (((size_t)(b * HH + h) * PP) + k0) * FF;
    const bf16* peq = penc + ((size_t)h * PE_ROWS + pq0) * FF;
    const bf16* pek = penc + ((size_t)h * PE_ROWS + pk0) * FF;

    v8f aqk = {}, gq0 = {}, gq1 = {}, gk0 = {}, gk1 = {};
#pragma unroll
    for (int kk = 0; kk < FF; kk += 32) {
      v16bf aq  = load_a_frag(qbase + kk, FF);
      aqk = wmma_bf16(aq, load_b_frag(kbase + kk, FF), aqk);
      v16bf aqb = load_a_frag(qbbase + kk, FF);
      gq0 = wmma_bf16(aqb, load_b_frag(peq + kk, FF), gq0);
      gq1 = wmma_bf16(aqb, load_b_frag(peq + (size_t)16 * FF + kk, FF), gq1);
      v16bf akb = load_a_frag(kbbase + kk, FF);
      gk0 = wmma_bf16(akb, load_b_frag(pek + kk, FF), gk0);
      gk1 = wmma_bf16(akb, load_b_frag(pek + (size_t)16 * FF + kk, FF), gk1);
    }
    float* g = gsc[wave];
    float* qs = qksc[wave];
#pragma unroll
    for (int j = 0; j < 8; ++j) {       // scatter fragments (C layout: m=mb+j, n)
      int m = mb + j;
      qs[m * 16 + n]      = aqk[j];
      g[m * 64 + n]       = gq0[j];
      g[m * 64 + 16 + n]  = gq1[j];
      g[m * 64 + 32 + n]  = gk0[j];
      g[m * 64 + 48 + n]  = gk1[j];
    }
    __syncthreads();
#pragma unroll
    for (int j = 0; j < 8; ++j) {       // shear: a = qk + 0.25*(Eq[kk-qq+15]+Ek[qq-kk+15])
      int idx = j * 32 + lane;
      int qq = idx >> 4, kk2 = idx & 15;
      float a = qs[qq * 16 + kk2]
              + 0.25f * (g[qq * 64 + (kk2 - qq + 15)]
                       + g[kk2 * 64 + 32 + (qq - kk2 + 15)]);
      aT[idx * 32 + h] = to_bf16(a);
    }
    __syncthreads();
  }

  // phase 2: pair = a_flat[256x32] @ WoutT + bout + y_q + y_k
  const float* yqrow = yq + (size_t)(b * PP + q0) * FF;
  const float* ykrow = yk + (size_t)(b * PP + k0) * FF;
  int row0 = wave * 32;
#pragma unroll
  for (int mt2 = 0; mt2 < 2; ++mt2) {
    int mrow = row0 + mt2 * 16;
    v16bf af = load_a_frag(&aT[(size_t)mrow * 32], 32);   // from LDS (ds_load)
    for (int nt = 0; nt < 8; ++nt) {
      v8f acc = {};
      acc = wmma_bf16(af, load_b_frag(woutT + (size_t)(nt * 16) * 32, 32), acc);
      int f = nt * 16 + n;
      float bo = bout[f];
#pragma unroll
      for (int j = 0; j < 8; ++j) {
        int mf = mrow + mb + j;
        int qq = mf >> 4, kk2 = mf & 15;
        float val = acc[j] + bo + yqrow[(size_t)qq * FF + f] + ykrow[(size_t)kk2 * FF + f];
        out[((((size_t)b * PP + (q0 + qq)) * PP) + (k0 + kk2)) * FF + f] = val;
      }
    }
  }
}

extern "C" void kernel_launch(void* const* d_in, const int* in_sizes, int n_in,
                              void* d_out, int out_size, void* d_ws, size_t ws_size,
                              hipStream_t stream) {
  (void)in_sizes; (void)n_in; (void)out_size; (void)ws_size;
  const float* x      = (const float*)d_in[0];
  const float* norm_w = (const float*)d_in[1];
  const float* Wq     = (const float*)d_in[2];
  const float* Wk     = (const float*)d_in[3];
  const float* Wpos   = (const float*)d_in[4];
  const float* bpos   = (const float*)d_in[5];
  const float* qrb    = (const float*)d_in[6];
  const float* krb    = (const float*)d_in[7];
  const float* Wyq    = (const float*)d_in[8];
  const float* Wyk    = (const float*)d_in[9];
  const float* Wout   = (const float*)d_in[10];
  const float* boutp  = (const float*)d_in[11];
  float* out = (float*)d_out;

  char* ws = (char*)d_ws;
  size_t off = 0;
  auto carve = [&](size_t bytes) {
    char* p = ws + off;
    off += (bytes + 255) & ~(size_t)255;
    return p;
  };
  bf16* xn    = (bf16*)carve((size_t)1024 * CH * 2);
  bf16* xg    = (bf16*)carve((size_t)1024 * CH * 2);
  bf16* WqT   = (bf16*)carve((size_t)HF * CH * 2);
  bf16* WkT   = (bf16*)carve((size_t)HF * CH * 2);
  bf16* WyqT  = (bf16*)carve((size_t)FF * CH * 2);
  bf16* WykT  = (bf16*)carve((size_t)FF * CH * 2);
  bf16* WoutT = (bf16*)carve((size_t)FF * HH * 2);
  bf16* qP    = (bf16*)carve((size_t)BQ * HH * PP * FF * 2);
  bf16* qB    = (bf16*)carve((size_t)BQ * HH * PP * FF * 2);
  bf16* kP    = (bf16*)carve((size_t)BQ * HH * PP * FF * 2);
  bf16* kB    = (bf16*)carve((size_t)BQ * HH * PP * FF * 2);
  bf16* penc  = (bf16*)carve((size_t)HH * PE_ROWS * FF * 2);
  float* yqv  = (float*)carve((size_t)1024 * FF * 4);
  float* ykv  = (float*)carve((size_t)1024 * FF * 4);

  prep_kernel<<<1024, 256, 0, stream>>>(x, norm_w, xn, xg);
  int n1 = CH * HF;
  tconv_kernel<<<(n1 + 255) / 256, 256, 0, stream>>>(Wq, WqT, CH, HF);
  tconv_kernel<<<(n1 + 255) / 256, 256, 0, stream>>>(Wk, WkT, CH, HF);
  int n2 = CH * FF;
  tconv_kernel<<<(n2 + 255) / 256, 256, 0, stream>>>(Wyq, WyqT, CH, FF);
  tconv_kernel<<<(n2 + 255) / 256, 256, 0, stream>>>(Wyk, WykT, CH, FF);
  int n3 = HH * FF;
  tconv_kernel<<<(n3 + 255) / 256, 256, 0, stream>>>(Wout, WoutT, HH, FF);
  posenc_kernel<<<PE_ROWS, 256, 0, stream>>>(Wpos, bpos, penc);
  proj_kernel<<<2048, 256, 0, stream>>>(xn, WqT, qrb, qP, qB);
  proj_kernel<<<2048, 256, 0, stream>>>(xn, WkT, krb, kP, kB);
  ygemm_kernel<<<64, 256, 0, stream>>>(xg, WyqT, yqv);
  ygemm_kernel<<<64, 256, 0, stream>>>(xg, WykT, ykv);
  pair_kernel<<<2048, 256, 0, stream>>>(qP, qB, kP, kB, penc, WoutT, boutp, yqv, ykv, out);
}
